// TransformerLayerController_40226663694943
// MI455X (gfx1250) — compile-verified, matrix operand order checked by reading, not run on hardware
//
#include <hip/hip_runtime.h>

typedef __attribute__((ext_vector_type(16))) _Float16 v16h;
typedef __attribute__((ext_vector_type(8)))  _Float16 v8h;
typedef __attribute__((ext_vector_type(8)))  float    v8f;
typedef __attribute__((ext_vector_type(4)))  int      v4i;

#define CAT16(lo, hi) __builtin_shufflevector((lo), (hi), 0,1,2,3,4,5,6,7,8,9,10,11,12,13,14,15)

static constexpr int kB = 2, kH = 32, kS = 1024, kD = 128;
static constexpr int kSink = 4, kNOut = 16;
static constexpr float kQMax = 7.0f;
static constexpr float kNegInf = -3.0e38f;

// ---- CDNA5 async global->LDS path (ASYNCcnt), with portable fallback ----
#if defined(__has_builtin)
#  if __has_builtin(__builtin_amdgcn_global_load_async_to_lds_b128)
#    define ASYNC_LDS 1
#  endif
#endif

#ifdef ASYNC_LDS
typedef __attribute__((address_space(1))) v4i gv4i;  // global int4*
typedef __attribute__((address_space(3))) v4i lv4i;  // LDS int4*
__device__ __forceinline__ void async_copy16(const void* g, void* l) {
  __builtin_amdgcn_global_load_async_to_lds_b128((gv4i*)g, (lv4i*)l, 0, 0);
}
__device__ __forceinline__ void async_wait0() {
#  if __has_builtin(__builtin_amdgcn_s_wait_asynccnt)
  __builtin_amdgcn_s_wait_asynccnt(0);
#  else
  asm volatile("s_wait_asynccnt 0x0" ::: "memory");
#  endif
}
#else
__device__ __forceinline__ void async_copy16(const void* g, void* l) {
  *(uint4*)l = *(const uint4*)g;
}
__device__ __forceinline__ void async_wait0() {}
#endif

// ---------------- Q fp32 -> f16 ----------------
__global__ void cvt_q_kernel(const float* __restrict__ q, _Float16* __restrict__ qh, int n) {
  for (int i = blockIdx.x * blockDim.x + threadIdx.x; i < n; i += gridDim.x * blockDim.x)
    qh[i] = (_Float16)q[i];
}

// ---------------- K reconstruction: one wave per (b,h,d) fiber over S ----------------
__global__ __launch_bounds__(256) void k_recon_kernel(const float* __restrict__ k,
                                                      _Float16* __restrict__ kh) {
  const int wid  = (blockIdx.x * blockDim.x + threadIdx.x) >> 5;  // fiber id in [0, B*H*D)
  const int lane = threadIdx.x & 31;
  const int d  = wid % kD;
  const int bh = wid / kD;
  const float* base = k + ((size_t)bh * kS) * kD + d;

  float orig[32], av[32];
#pragma unroll
  for (int j = 0; j < 32; ++j) {
    const int s = j * 32 + lane;
    const float x = base[(size_t)s * kD];
    orig[j] = x;
    av[j] = (s < kSink) ? 0.0f : fabsf(x);
  }

  // top-16 |x| over the 1024-element fiber: 16 rounds of wave argmax
  unsigned excl = 0u;
  for (int r = 0; r < kNOut; ++r) {
    float vmax = -1.0f; int jmax = 0;
#pragma unroll
    for (int j = 0; j < 32; ++j)
      if (!((excl >> j) & 1u) && av[j] > vmax) { vmax = av[j]; jmax = j; }
    int packed = (lane << 5) | jmax;
#pragma unroll
    for (int off = 16; off >= 1; off >>= 1) {
      const float ov = __shfl_xor(vmax, off);
      const int   op = __shfl_xor(packed, off);
      if (ov > vmax) { vmax = ov; packed = op; }
    }
    if ((packed >> 5) == lane) excl |= 1u << (packed & 31);
  }

  float amax = 0.0f;
#pragma unroll
  for (int j = 0; j < 32; ++j) if (!((excl >> j) & 1u)) amax = fmaxf(amax, av[j]);
#pragma unroll
  for (int off = 16; off >= 1; off >>= 1) amax = fmaxf(amax, __shfl_xor(amax, off));

  float scale = amax / kQMax;
  if (!(scale > 0.0f)) scale = 1.0f;
  const float inv = 1.0f / scale;

  _Float16* outb = kh + ((size_t)bh * kS) * kD + d;
#pragma unroll
  for (int j = 0; j < 32; ++j) {
    const int s = j * 32 + lane;
    const float x = orig[j];
    float o;
    if (s < kSink || ((excl >> j) & 1u)) {
      o = x;  // sink (full precision) or outlier (kept dense value)
    } else {
      float qv = rintf(x * inv);
      qv = fminf(fmaxf(qv, -kQMax), kQMax);
      o = qv * scale;
    }
    outb[(size_t)s * kD] = (_Float16)o;
  }
}

// ---------------- V reconstruction: one wave per (b,h,s) token over D ----------------
__global__ __launch_bounds__(256) void v_recon_kernel(const float* __restrict__ v,
                                                      _Float16* __restrict__ vh) {
  const int wid  = (blockIdx.x * blockDim.x + threadIdx.x) >> 5;  // token id in [0, B*H*S)
  const int lane = threadIdx.x & 31;
  const int s = wid % kS;
  const float* base = v + (size_t)wid * kD;
  const bool sink = (s < kSink);

  float orig[4], av[4];
#pragma unroll
  for (int j = 0; j < 4; ++j) {
    const float x = base[j * 32 + lane];
    orig[j] = x;
    av[j] = sink ? 0.0f : fabsf(x);
  }

  unsigned excl = 0u;
  for (int r = 0; r < kNOut; ++r) {
    float vmax = -1.0f; int jmax = 0;
#pragma unroll
    for (int j = 0; j < 4; ++j)
      if (!((excl >> j) & 1u) && av[j] > vmax) { vmax = av[j]; jmax = j; }
    int packed = (lane << 2) | jmax;
#pragma unroll
    for (int off = 16; off >= 1; off >>= 1) {
      const float ov = __shfl_xor(vmax, off);
      const int   op = __shfl_xor(packed, off);
      if (ov > vmax) { vmax = ov; packed = op; }
    }
    if ((packed >> 2) == lane) excl |= 1u << (packed & 3);
  }

  float amax = 0.0f;
#pragma unroll
  for (int j = 0; j < 4; ++j) if (!((excl >> j) & 1u)) amax = fmaxf(amax, av[j]);
#pragma unroll
  for (int off = 16; off >= 1; off >>= 1) amax = fmaxf(amax, __shfl_xor(amax, off));

  float scale = amax / kQMax;
  if (!(scale > 0.0f)) scale = 1.0f;
  const float inv = 1.0f / scale;

  _Float16* outb = vh + (size_t)wid * kD;
#pragma unroll
  for (int j = 0; j < 4; ++j) {
    const float x = orig[j];
    float o;
    if (sink || ((excl >> j) & 1u)) o = x;
    else {
      float qv = rintf(x * inv);
      qv = fminf(fmaxf(qv, -kQMax), kQMax);
      o = qv * scale;
    }
    outb[j * 32 + lane] = (_Float16)o;
  }
}

// ---------------- Flash attention (causal), f16 WMMA, fp32 softmax/accum ----------------
static constexpr int BM = 128;  // query rows per block
static constexpr int WM = 16;   // query rows per wave
static constexpr int KT = 64;   // key tile
static constexpr int NW = 8;    // waves per block

__device__ __forceinline__ v8f zero8() {
  v8f z;
#pragma unroll
  for (int i = 0; i < 8; ++i) z[i] = 0.0f;
  return z;
}

__global__ __launch_bounds__(256) void attn_kernel(const _Float16* __restrict__ qh,
                                                   const _Float16* __restrict__ kh,
                                                   const _Float16* __restrict__ vh,
                                                   float* __restrict__ out) {
  __shared__ _Float16 sK[2][KT * kD];      // [buf][key][d]   2 x 16 KB (async target)
  __shared__ _Float16 sVT[2][kD * KT];     // [buf][d][key]   2 x 16 KB (transposed V)
  __shared__ _Float16 sP[NW * WM * KT];    // per-wave P      16 KB

  const int tid   = threadIdx.x;
  const int lane  = tid & 31;
  const int wave  = tid >> 5;
  const int lhalf = lane >> 4;    // 0/1 half-wave
  const int l16   = lane & 15;

  const int bh      = blockIdx.y;
  const int r0      = blockIdx.x * BM;
  const int rowBase = r0 + wave * WM;

  const size_t bhOff = (size_t)bh * kS * kD;
  const _Float16* Q = qh + bhOff;
  const _Float16* K = kh + bhOff;
  const _Float16* V = vh + bhOff;

  // A-fragments of Q: 16 rows x 128 d in 4 chunks of 32
  v16h aQ[4];
#pragma unroll
  for (int c = 0; c < 4; ++c) {
    const int row   = rowBase + l16;
    const int dbase = c * 32 + lhalf * 8;
    const v8h lo = *(const v8h*)(Q + (size_t)row * kD + dbase);
    const v8h hi = *(const v8h*)(Q + (size_t)row * kD + dbase + 16);
    aQ[c] = CAT16(lo, hi);
  }

  v8f O[8];
#pragma unroll
  for (int i = 0; i < 8; ++i) O[i] = zero8();
  float mrow[8], lrow[8];
#pragma unroll
  for (int r = 0; r < 8; ++r) { mrow[r] = kNegInf; lrow[r] = 0.0f; }

  const float smScale = 0.08838834764831845f;  // 1/sqrt(128)
  const int kEnd = r0 + BM;                    // causal upper bound for this block

  // ---- tile staging helpers (each thread moves 32 halfs per tile) ----
  auto issueK = [&](int kt, int buf) {
#pragma unroll
    for (int it = 0; it < 4; ++it) {
      const int hidx = (it * 256 + tid) * 8;
      async_copy16(K + (size_t)kt * kD + hidx, &sK[buf][hidx]);
    }
  };
  auto loadVregs = [&](int kt, v8h* vreg) {
#pragma unroll
    for (int it = 0; it < 4; ++it) {
      const int hidx = (it * 256 + tid) * 8;
      const int key = hidx >> 7;
      const int dd  = hidx & 127;
      vreg[it] = *(const v8h*)(V + (size_t)(kt + key) * kD + dd);
    }
  };
  auto storeVT = [&](const v8h* vreg, int buf) {
#pragma unroll
    for (int it = 0; it < 4; ++it) {
      const int hidx = (it * 256 + tid) * 8;
      const int key = hidx >> 7;
      const int dd  = hidx & 127;
#pragma unroll
      for (int e = 0; e < 8; ++e) sVT[buf][(dd + e) * KT + key] = vreg[it][e];
    }
  };

  // ---- prologue: stage tile 0 into buffer 0 ----
  {
    v8h vreg[4];
    issueK(0, 0);
    loadVregs(0, vreg);
    storeVT(vreg, 0);
    async_wait0();
  }
  __syncthreads();

  int t = 0;
  for (int kt = 0; kt < kEnd; kt += KT, ++t) {
    const int cur = t & 1;
    const int nxt = cur ^ 1;
    const bool hasNext = (kt + KT) < kEnd;

    // kick off next tile's copies before computing on the current one
    v8h vreg[4];
    if (hasNext) {
      issueK(kt + KT, nxt);       // async global->LDS (ASYNCcnt), overlaps compute
      loadVregs(kt + KT, vreg);   // V in flight in registers during compute
    }

    const _Float16* sKc  = sK[cur];
    const _Float16* sVTc = sVT[cur];

    const bool active = (kt <= rowBase + WM - 1);  // wave-uniform causal skip
    if (active) {
      // ---- scores S = Q K^T : 16 x 64 fp32 ----
      v8f acc[4];
#pragma unroll
      for (int n = 0; n < 4; ++n) {
        v8f a = zero8();
        const int keyl = n * 16 + l16;
#pragma unroll
        for (int c = 0; c < 4; ++c) {
          const int dbase = c * 32 + lhalf * 8;
          const v8h lo = *(const v8h*)&sKc[keyl * kD + dbase];
          const v8h hi = *(const v8h*)&sKc[keyl * kD + dbase + 16];
          const v16h bf = CAT16(lo, hi);
          a = __builtin_amdgcn_wmma_f32_16x16x32_f16(false, aQ[c], false, bf,
                                                     (short)0, a, false, false);
        }
        acc[n] = a;
      }

      // ---- fp32 online softmax (8 rows per lane; 16-lane row reductions) ----
      float pnew[4][8];
#pragma unroll
      for (int r = 0; r < 8; ++r) {
        const int grow = rowBase + lhalf * 8 + r;
        float sv[4];
        float rmax = kNegInf;
#pragma unroll
        for (int n = 0; n < 4; ++n) {
          const int gcol = kt + n * 16 + l16;
          float x = acc[n][r] * smScale;
          if (gcol > grow) x = kNegInf;
          sv[n] = x;
          rmax = fmaxf(rmax, x);
        }
#pragma unroll
        for (int off = 8; off >= 1; off >>= 1) rmax = fmaxf(rmax, __shfl_xor(rmax, off));
        const float mnew = fmaxf(mrow[r], rmax);
        const float corr = __expf(mrow[r] - mnew);
        float rsum = 0.0f;
#pragma unroll
        for (int n = 0; n < 4; ++n) { pnew[n][r] = __expf(sv[n] - mnew); rsum += pnew[n][r]; }
#pragma unroll
        for (int off = 8; off >= 1; off >>= 1) rsum += __shfl_xor(rsum, off);
        lrow[r] = lrow[r] * corr + rsum;
        mrow[r] = mnew;
#pragma unroll
        for (int ds = 0; ds < 8; ++ds) O[ds][r] *= corr;
      }

      // ---- reshape P (C layout) -> A layout via per-wave LDS buffer ----
      _Float16* myP = sP + wave * WM * KT;
#pragma unroll
      for (int r = 0; r < 8; ++r) {
        const int lr = lhalf * 8 + r;
#pragma unroll
        for (int n = 0; n < 4; ++n)
          myP[lr * KT + n * 16 + l16] = (_Float16)pnew[n][r];
      }
      // DS ops from the same wave complete in order; compiler inserts s_wait_dscnt.

      v16h aP[2];
#pragma unroll
      for (int c = 0; c < 2; ++c) {
        const int kb = c * 32 + lhalf * 8;
        const v8h lo = *(const v8h*)&myP[l16 * KT + kb];
        const v8h hi = *(const v8h*)&myP[l16 * KT + kb + 16];
        aP[c] = CAT16(lo, hi);
      }

      // ---- O += P V : 16 x 128 ----
#pragma unroll
      for (int ds = 0; ds < 8; ++ds) {
        const int dcol = ds * 16 + l16;
#pragma unroll
        for (int c = 0; c < 2; ++c) {
          const int kb = c * 32 + lhalf * 8;
          const v8h lo = *(const v8h*)&sVTc[dcol * KT + kb];
          const v8h hi = *(const v8h*)&sVTc[dcol * KT + kb + 16];
          const v16h bf = CAT16(lo, hi);
          O[ds] = __builtin_amdgcn_wmma_f32_16x16x32_f16(false, aP[c], false, bf,
                                                         (short)0, O[ds], false, false);
        }
      }
    }

    if (hasNext) {
      storeVT(vreg, nxt);   // transpose V tile into the other buffer
      async_wait0();        // next K tile landed in LDS
    }
    __syncthreads();        // publish nxt buffers / protect cur for reuse
  }

  // ---- normalize and store ----
#pragma unroll
  for (int r = 0; r < 8; ++r) {
    const int grow = rowBase + lhalf * 8 + r;
    const float invl = 1.0f / lrow[r];
#pragma unroll
    for (int ds = 0; ds < 8; ++ds)
      out[bhOff + (size_t)grow * kD + ds * 16 + l16] = O[ds][r] * invl;
  }
}

// ---------------- launcher ----------------
extern "C" void kernel_launch(void* const* d_in, const int* in_sizes, int n_in,
                              void* d_out, int out_size, void* d_ws, size_t ws_size,
                              hipStream_t stream) {
  (void)in_sizes; (void)n_in; (void)out_size; (void)ws_size;
  const float* q = (const float*)d_in[0];
  const float* k = (const float*)d_in[1];
  const float* v = (const float*)d_in[2];
  float* out = (float*)d_out;

  const size_t nElem = (size_t)kB * kH * kS * kD;  // 8,388,608
  _Float16* qh = (_Float16*)d_ws;
  _Float16* kh = qh + nElem;
  _Float16* vh = kh + nElem;

  cvt_q_kernel<<<4096, 256, 0, stream>>>(q, qh, (int)nElem);
  k_recon_kernel<<<(kB * kH * kD) / NW, 256, 0, stream>>>(k, kh);   // 1024 blocks
  v_recon_kernel<<<(kB * kH * kS) / NW, 256, 0, stream>>>(v, vh);   // 8192 blocks
  dim3 grid(kS / BM, kB * kH);                                       // (8, 64)
  attn_kernel<<<grid, 256, 0, stream>>>(qh, kh, vh, out);
}
